// DCGRUCell_56779467653495
// MI455X (gfx1250) — compile-verified, matrix operand order checked by reading, not run on hardware
//
#include <hip/hip_runtime.h>
#include <hip/hip_bf16.h>

// ---------------- Problem constants (from reference) ----------------
#define NNODES   10000
#define UDIM     64
#define DIN      2
#define BSZ      32
#define FDIM     66                 // DIN + UDIM
#define NMATS    5                  // 2 supports * K(=2) + identity
#define KTOT     330                // NMATS * FDIM
#define KPAD     352                // KTOT rounded up to multiple of 32
#define FB       (FDIM * BSZ)       // 2112 : width of x0 rows
#define NNZ_E    200000
#define ROWS     (BSZ * NNODES)     // 320000 GEMM rows (N % 16 == 0 -> tiles never cross batch)
#define BNU      (BSZ * NNODES * UDIM)

typedef __attribute__((ext_vector_type(16))) _Float16 v16h;
typedef __attribute__((ext_vector_type(8)))  _Float16 v8h;
typedef __attribute__((ext_vector_type(8)))  float    v8f;

// ---------------- feature assembly ----------------
// x0[n, b, f] = concat(inputs, state_like)[b, n, f]
__global__ __launch_bounds__(256) void build_x0_kernel(
    const float* __restrict__ inp, const float* __restrict__ st, float* __restrict__ x0)
{
    int idx = blockIdx.x * 256 + threadIdx.x;           // N*B*F = 21,120,000 total
    if (idx >= NNODES * BSZ * FDIM) return;
    int f = idx % FDIM;
    int nb = idx / FDIM;
    int b = nb & (BSZ - 1);
    int n = nb >> 5;
    float v = (f < DIN) ? inp[b * (NNODES * DIN) + n * DIN + f]
                        : st [b * (NNODES * UDIM) + n * UDIM + (f - DIN)];
    x0[idx] = v;
}

// ---------------- on-device CSR build (per support) ----------------
__global__ __launch_bounds__(256) void zero_int_kernel(int* __restrict__ p, int n)
{
    int i = blockIdx.x * 256 + threadIdx.x;
    if (i < n) p[i] = 0;
}

__global__ __launch_bounds__(256) void hist_kernel(
    const int* __restrict__ rows, int* __restrict__ cnt)
{
    int e = blockIdx.x * 256 + threadIdx.x;
    if (e < NNZ_E) atomicAdd(&cnt[rows[e]], 1);
}

// single-block exclusive scan over row counts -> rowptr, and copy to cursor
#define SCAN_T 1024
#define SCAN_CHUNK 10               // 1024*10 = 10240 >= NNODES
__global__ __launch_bounds__(SCAN_T) void scan_kernel(
    const int* __restrict__ cnt, int* __restrict__ rowptr, int* __restrict__ cursor)
{
    __shared__ int sh[SCAN_T];
    int t = threadIdx.x;
    int base = t * SCAN_CHUNK;
    int loc[SCAN_CHUNK];
    int s = 0;
#pragma unroll
    for (int i = 0; i < SCAN_CHUNK; ++i) {
        int r = base + i;
        int v = (r < NNODES) ? cnt[r] : 0;
        loc[i] = s;
        s += v;
    }
    sh[t] = s;
    __syncthreads();
    for (int off = 1; off < SCAN_T; off <<= 1) {
        int v = sh[t];
        int u = (t >= off) ? sh[t - off] : 0;
        __syncthreads();
        sh[t] = v + u;
        __syncthreads();
    }
    int excl = sh[t] - s;
#pragma unroll
    for (int i = 0; i < SCAN_CHUNK; ++i) {
        int r = base + i;
        if (r < NNODES) {
            int p = excl + loc[i];
            rowptr[r] = p;
            cursor[r] = p;
        }
    }
    if (t == SCAN_T - 1) rowptr[NNODES] = sh[t];
}

__global__ __launch_bounds__(256) void scatter_kernel(
    const int* __restrict__ rows, const int* __restrict__ cols, const float* __restrict__ vals,
    int* __restrict__ cursor, int* __restrict__ cols_csr, float* __restrict__ vals_csr)
{
    int e = blockIdx.x * 256 + threadIdx.x;
    if (e >= NNZ_E) return;
    int r = rows[e];
    int pos = atomicAdd(&cursor[r], 1);
    cols_csr[pos] = cols[e];
    vals_csr[pos] = vals[e];
}

// ---------------- atomic-free row-parallel SpMM ----------------
// one block per row; y[r,:] = alpha * (S @ x)[r,:] - (sub ? sub[r,:] : 0)
__global__ __launch_bounds__(256) void spmm_csr_kernel(
    const int* __restrict__ rowptr, const int* __restrict__ cols_csr,
    const float* __restrict__ vals_csr, const float* __restrict__ x,
    const float* __restrict__ sub, float* __restrict__ y, float alpha)
{
    int r = blockIdx.x;
    int t = threadIdx.x;
    int e0 = rowptr[r], e1 = rowptr[r + 1];
    const bool extra = (t < FB - 8 * 256);              // last 64 cols
    float acc[9];
#pragma unroll
    for (int i = 0; i < 9; ++i) acc[i] = 0.f;
    for (int e = e0; e < e1; ++e) {
        int c = cols_csr[e];
        float v = vals_csr[e];
        const float* xp = x + (size_t)c * FB;
#pragma unroll
        for (int i = 0; i < 8; ++i) acc[i] += v * xp[t + i * 256];
        if (extra) acc[8] += v * xp[t + 2048];
    }
    float* yp = y + (size_t)r * FB;
    if (sub) {
        const float* sp = sub + (size_t)r * FB;
#pragma unroll
        for (int i = 0; i < 8; ++i) yp[t + i * 256] = alpha * acc[i] - sp[t + i * 256];
        if (extra) yp[t + 2048] = alpha * acc[8] - sp[t + 2048];
    } else {
#pragma unroll
        for (int i = 0; i < 8; ++i) yp[t + i * 256] = alpha * acc[i];
        if (extra) yp[t + 2048] = alpha * acc[8];
    }
}

// Wp[m*66+f, c] = W[f*5+m, c]  (regroup K so each diffusion matrix is contiguous)
__global__ __launch_bounds__(256) void permute_w_kernel(
    const float* __restrict__ w, float* __restrict__ wp, int outDim)
{
    int idx = blockIdx.x * 256 + threadIdx.x;
    if (idx >= KTOT * outDim) return;
    int c = idx % outDim;
    int k = idx / outDim;
    int f = k / NMATS;
    int m = k - f * NMATS;
    wp[(m * FDIM + f) * outDim + c] = w[idx];
}

__device__ __forceinline__ float sigmoidf_(float x) { return 1.f / (1.f + __expf(-x)); }

// ---------------- software-pipelined WMMA GEMM with fused GRU epilogues ----------------
// Virtual A(row = b*N+n, k' = m*66+f) = xs[m][(n*B + b)*66 + f]
// GATE=true  (OUT=128): tiles 0..3 -> rstate = sigmoid(r)*state ; tiles 4..7 -> ugate = sigmoid(u)
// GATE=false (OUT=64) : out/new_state = ugate*state + (1-ugate)*tanh(cand)
template<int OUT, bool GATE>
__global__ __launch_bounds__(256) void gconv_gemm_wmma_kernel(
    const float* __restrict__ x0, const float* __restrict__ x1,
    const float* __restrict__ x2, const float* __restrict__ x3,
    const float* __restrict__ x4,
    const float* __restrict__ wp, const float* __restrict__ bias,
    const float* __restrict__ state, const float* __restrict__ ugate,
    float* __restrict__ outA, float* __restrict__ outB)
{
    constexpr int NT = OUT / 16;
    constexpr int NCHUNK = KPAD / 32;                   // 11
    constexpr int NB = (32 * OUT) / 256;                // B elements per thread
    constexpr int BKSTEP = 256 / OUT;
    __shared__ alignas(16) _Float16 As[2][128][40];     // ping-pong, pitch 40 halves
    __shared__ alignas(16) _Float16 Bs[2][OUT][40];     // B^T

    const int row0 = blockIdx.x * 128;
    const int t    = threadIdx.x;
    const int wv   = t >> 5;
    const int lane = t & 31;
    const int lrow = lane & 15;
    const int koff = (lane >> 4) * 8;

    const int akk  = t & 31;                            // fixed k within chunk
    const int arr0 = t >> 5;                            // row stride 8
    const int bcc  = t % OUT;
    const int bkk0 = t / OUT;

    float areg[16];
    float breg[NB];

    auto gload = [&](int kc) {
        int k = kc + akk;
        bool ok = (k < KTOT);
        const float* base = nullptr;
        int f = 0;
        if (ok) {
            int m = k / FDIM;
            f = k - m * FDIM;
            base = (m == 0) ? x0 : (m == 1) ? x1 : (m == 2) ? x2 : (m == 3) ? x3 : x4;
        }
#pragma unroll
        for (int i = 0; i < 16; ++i) {
            float a = 0.f;
            if (ok) {
                int row = row0 + arr0 + i * 8;
                int b = row / NNODES;
                int n = row - b * NNODES;
                a = base[(size_t)(n * BSZ + b) * FDIM + f];
            }
            areg[i] = a;
        }
#pragma unroll
        for (int i = 0; i < NB; ++i) {
            int k2 = kc + bkk0 + i * BKSTEP;
            breg[i] = (k2 < KTOT) ? wp[k2 * OUT + bcc] : 0.f;
        }
    };
    auto lstore = [&](int buf) {
#pragma unroll
        for (int i = 0; i < 16; ++i) As[buf][arr0 + i * 8][akk] = (_Float16)areg[i];
#pragma unroll
        for (int i = 0; i < NB; ++i) Bs[buf][bcc][bkk0 + i * BKSTEP] = (_Float16)breg[i];
    };

    v8f acc[NT];
#pragma unroll
    for (int i = 0; i < NT; ++i)
#pragma unroll
        for (int g = 0; g < 8; ++g) acc[i][g] = 0.f;

    gload(0);
    lstore(0);

    for (int c = 0; c < NCHUNK; ++c) {
        __syncthreads();
        int cur = c & 1;
        if (c + 1 < NCHUNK) gload((c + 1) * 32);        // prefetch next chunk into regs

        const _Float16* ap = &As[cur][wv * 16 + lrow][0];
        v8h alo = *(const v8h*)(ap + koff);
        v8h ahi = *(const v8h*)(ap + koff + 16);
        v16h af;
#pragma unroll
        for (int i = 0; i < 8; ++i) { af[i] = alo[i]; af[i + 8] = ahi[i]; }

#pragma unroll
        for (int ct = 0; ct < NT; ++ct) {
            const _Float16* bp = &Bs[cur][ct * 16 + lrow][0];
            v8h blo = *(const v8h*)(bp + koff);
            v8h bhi = *(const v8h*)(bp + koff + 16);
            v16h bf;
#pragma unroll
            for (int i = 0; i < 8; ++i) { bf[i] = blo[i]; bf[i + 8] = bhi[i]; }
            acc[ct] = __builtin_amdgcn_wmma_f32_16x16x32_f16(
                false, af, false, bf, (short)0, acc[ct], false, false);
        }
        if (c + 1 < NCHUNK) lstore((c + 1) & 1);
    }

    // ---- fused epilogue (C layout: VGPR g -> M = g + 8*(lane>=16), N = lane&15) ----
#pragma unroll
    for (int ct = 0; ct < NT; ++ct) {
        int col = ct * 16 + lrow;
        float bv = bias[col];
#pragma unroll
        for (int g = 0; g < 8; ++g) {
            int m = (lane >> 4) * 8 + g;
            int grow = row0 + wv * 16 + m;
            float val = acc[ct][g] + bv;
            if constexpr (GATE) {
                float sig = sigmoidf_(val);
                if (col < UDIM) {                        // r-gate -> rstate
                    size_t idx = (size_t)grow * UDIM + col;
                    outA[idx] = sig * state[idx];        // outA = rstate
                } else {                                 // u-gate -> sigmoid stored
                    size_t idx = (size_t)grow * UDIM + (col - UDIM);
                    outB[idx] = sig;                     // outB = ugate
                }
            } else {
                size_t idx = (size_t)grow * UDIM + col;
                float ug = ugate[idx];
                float ns = ug * state[idx] + (1.f - ug) * tanhf(val);
                outA[idx] = ns;                          // output
                outB[idx] = ns;                          // new_state (tuple duplicate)
            }
        }
    }
}

// ---------------- host-side orchestration ----------------
static void build_csr(const int* rows, const int* cols, const float* vals,
                      int* rowptr, int* cursor, int* cols_csr, float* vals_csr,
                      hipStream_t stream)
{
    const int eGrid = (NNZ_E + 255) / 256;
    zero_int_kernel<<<(NNODES + 255) / 256, 256, 0, stream>>>(cursor, NNODES);
    hist_kernel<<<eGrid, 256, 0, stream>>>(rows, cursor);
    scan_kernel<<<1, SCAN_T, 0, stream>>>(cursor, rowptr, cursor);
    scatter_kernel<<<eGrid, 256, 0, stream>>>(rows, cols, vals, cursor, cols_csr, vals_csr);
}

static void run_gconv_spmms(const float* x0, float* xs1, float* xs2, float* xs3, float* xs4,
                            const int* rp0, const int* cc0, const float* vv0,
                            const int* rp1, const int* cc1, const float* vv1,
                            hipStream_t stream)
{
    // support 0: x1 = S0 @ x0 ; x2 = 2*(S0 @ x1) - x0
    spmm_csr_kernel<<<NNODES, 256, 0, stream>>>(rp0, cc0, vv0, x0, nullptr, xs1, 1.f);
    spmm_csr_kernel<<<NNODES, 256, 0, stream>>>(rp0, cc0, vv0, xs1, x0, xs2, 2.f);
    // support 1
    spmm_csr_kernel<<<NNODES, 256, 0, stream>>>(rp1, cc1, vv1, x0, nullptr, xs3, 1.f);
    spmm_csr_kernel<<<NNODES, 256, 0, stream>>>(rp1, cc1, vv1, xs3, x0, xs4, 2.f);
}

extern "C" void kernel_launch(void* const* d_in, const int* in_sizes, int n_in,
                              void* d_out, int out_size, void* d_ws, size_t ws_size,
                              hipStream_t stream)
{
    const float* inputs  = (const float*)d_in[0];
    const float* state   = (const float*)d_in[1];
    const int*   s0_rows = (const int*)  d_in[2];
    const int*   s0_cols = (const int*)  d_in[3];
    const float* s0_vals = (const float*)d_in[4];
    const int*   s1_rows = (const int*)  d_in[5];
    const int*   s1_cols = (const int*)  d_in[6];
    const float* s1_vals = (const float*)d_in[7];
    const float* gate_w  = (const float*)d_in[8];
    const float* gate_b  = (const float*)d_in[9];
    const float* cand_w  = (const float*)d_in[10];
    const float* cand_b  = (const float*)d_in[11];
    float* out = (float*)d_out;

    // workspace layout (floats, then ints)
    float* ws = (float*)d_ws;
    const size_t XSZ = (size_t)NNODES * BSZ * FDIM;     // 21,120,000
    float* xs0 = ws;
    float* xs1 = xs0 + XSZ;
    float* xs2 = xs1 + XSZ;
    float* xs3 = xs2 + XSZ;
    float* xs4 = xs3 + XSZ;
    float* rstate = xs4 + XSZ;                          // BNU
    float* ugate  = rstate + (size_t)BNU;               // BNU
    float* wp_g   = ugate + (size_t)BNU;                // 330*128
    float* wp_c   = wp_g + (size_t)KTOT * 128;          // 330*64
    float* v0_csr = wp_c + (size_t)KTOT * 64;           // NNZ
    float* v1_csr = v0_csr + (size_t)NNZ_E;             // NNZ
    int*   ip     = (int*)(v1_csr + (size_t)NNZ_E);
    int*   rp0    = ip;                                 // NNODES+1
    int*   rp1    = rp0 + (NNODES + 1);                 // NNODES+1
    int*   cur0   = rp1 + (NNODES + 1);                 // NNODES
    int*   cur1   = cur0 + NNODES;                      // NNODES
    int*   c0_csr = cur1 + NNODES;                      // NNZ
    int*   c1_csr = c0_csr + NNZ_E;                     // NNZ

    // weights: regroup K to k' = m*66+f
    permute_w_kernel<<<(KTOT * 128 + 255) / 256, 256, 0, stream>>>(gate_w, wp_g, 128);
    permute_w_kernel<<<(KTOT * 64 + 255) / 256, 256, 0, stream>>>(cand_w, wp_c, 64);

    // CSR for both supports (built once, used by 8 SpMMs)
    build_csr(s0_rows, s0_cols, s0_vals, rp0, cur0, c0_csr, v0_csr, stream);
    build_csr(s1_rows, s1_cols, s1_vals, rp1, cur1, c1_csr, v1_csr, stream);

    const int x0Grid = (NNODES * BSZ * FDIM + 255) / 256;

    // ===== gconv #1 (gate) -> fused sigmoid epilogue writes rstate & ugate =====
    build_x0_kernel<<<x0Grid, 256, 0, stream>>>(inputs, state, xs0);
    run_gconv_spmms(xs0, xs1, xs2, xs3, xs4, rp0, c0_csr, v0_csr, rp1, c1_csr, v1_csr, stream);
    gconv_gemm_wmma_kernel<128, true><<<ROWS / 128, 256, 0, stream>>>(
        xs0, xs1, xs2, xs3, xs4, wp_g, gate_b, state, nullptr, rstate, ugate);

    // ===== gconv #2 (candidate) -> fused tanh + GRU combine writes both outputs =====
    build_x0_kernel<<<x0Grid, 256, 0, stream>>>(inputs, rstate, xs0);
    run_gconv_spmms(xs0, xs1, xs2, xs3, xs4, rp0, c0_csr, v0_csr, rp1, c1_csr, v1_csr, stream);
    gconv_gemm_wmma_kernel<64, false><<<ROWS / 128, 256, 0, stream>>>(
        xs0, xs1, xs2, xs3, xs4, wp_c, cand_b, state, ugate, out, out + (size_t)BNU);
}
